// LSTM_14568529068032
// MI455X (gfx1250) — compile-verified, matrix operand order checked by reading, not run on hardware
//
#include <hip/hip_runtime.h>
#include <hip/hip_bf16.h>

typedef __attribute__((ext_vector_type(16))) _Float16 v16h;
typedef __attribute__((ext_vector_type(8)))  _Float16 v8h;
typedef __attribute__((ext_vector_type(8)))  float    v8f;
typedef __attribute__((ext_vector_type(4)))  unsigned int v4u;
typedef __attribute__((ext_vector_type(8)))  int      v8i;
typedef __attribute__((ext_vector_type(4)))  int      v4i;

#define B_  256
#define V_  100
#define C_  40
#define E_  128
#define H_  100
#define KP_ 128
#define NJT 28   // 4 gates * 7 column tiles (H padded 100->112)
#define NMT 8    // 128 rows / 16

// ---- workspace layout (bytes) ----
static constexpr size_t OFF_X    = 0;                       // f16 x[t=256][m=128][k=128]
static constexpr size_t SZ_X     = 256ull*128*128*2;        // 8,388,608
static constexpr size_t OFF_WIH  = OFF_X + SZ_X;            // f16 fragments, 112 frags * 1KB
static constexpr size_t SZ_WF    = 112ull*1024;             // 114,688
static constexpr size_t OFF_WHH  = OFF_WIH + SZ_WF;
static constexpr size_t OFF_LAST = OFF_WHH + SZ_WF;         // int last[256]
static constexpr size_t OFF_XG   = OFF_LAST + 1024;         // f32 Xg tiles: 256*8*28 * 1KB

__device__ __forceinline__ float fsig(float x) {
    x = fminf(fmaxf(x, -30.f), 30.f);
    return 1.f / (1.f + __expf(-x));
}
__device__ __forceinline__ float ftanh(float x) {
    x = fminf(fmaxf(x, -15.f), 15.f);
    float e = __expf(2.f * x);
    return (e - 1.f) / (e + 1.f);
}
__device__ __forceinline__ v8f v8f_splat(float s) {
    v8f r;
#pragma unroll
    for (int i = 0; i < 8; ++i) r[i] = s;
    return r;
}

union AB { v16h v; v8h h[2]; };

// ---- TDM flat copy: global f32 buffer -> LDS (one descriptor, 2D degenerate tile)
__device__ __forceinline__ void tdm_copy_to_lds(const void* gsrc, void* lds_generic,
                                                unsigned nelem_f32) {
#if __has_builtin(__builtin_amdgcn_tensor_load_to_lds) && __has_builtin(__builtin_amdgcn_s_wait_tensorcnt)
    unsigned long long ga = (unsigned long long)gsrc;
    unsigned lga = (unsigned)(unsigned long long)lds_generic;  // LDS aperture: low 32 bits = offset
    // D# group 0: count=1 | lds_addr | global_addr[56:0] | type=2
    v4u g0;
    g0[0] = 1u;                                   // count=1, user mode
    g0[1] = lga;                                  // lds_addr
    g0[2] = (unsigned)(ga & 0xffffffffu);         // global_addr[31:0]
    g0[3] = (unsigned)((ga >> 32) & 0x01ffffffu)  // global_addr[56:32]
          | (2u << 30);                           // type=2 ("image")
    // D# group 1: data_size=4B, tensor_dim0=tile_dim0=N, dims1=1, stride0=N
    v8i g1;
    g1[0] = (int)(2u << 16);                      // data_size=2 -> 4 bytes; mask=0
    g1[1] = (int)((nelem_f32 & 0xffffu) << 16);   // tensor_dim0[15:0] in bits 63:48
    g1[2] = (int)(((nelem_f32 >> 16) & 0xffffu)   // tensor_dim0[31:16]
          | (1u << 16));                          // tensor_dim1 = 1 (low 16)
    g1[3] = (int)((nelem_f32 & 0xffffu) << 16);   // tensor_dim1 hi=0 | tile_dim0[15:0]
    g1[4] = 1;                                    // tile_dim1=1 | tile_dim2=0
    g1[5] = (int)nelem_f32;                       // tensor_dim0_stride[31:0]
    g1[6] = 0;                                    // stride0 hi | stride1 lo
    g1[7] = 0;
    v4i gz = { 0, 0, 0, 0 };
#if __clang_major__ >= 23
    v8i gz8 = { 0, 0, 0, 0, 0, 0, 0, 0 };
    __builtin_amdgcn_tensor_load_to_lds(g0, g1, gz, gz, gz8, 0);
#else
    __builtin_amdgcn_tensor_load_to_lds(g0, g1, gz, gz, 0);
#endif
    __builtin_amdgcn_s_wait_tensorcnt(0);
#else
    // fallback: cooperative copy (wave 0 only reaches here; do full copy serially per lane)
    const float* s = (const float*)gsrc;
    float* d = (float*)lds_generic;
    for (unsigned i = (unsigned)threadIdx.x; i < nelem_f32; i += 32u) d[i] = s[i];
#endif
}

// ---- init: zero x (for k-padding) and last[] ----
__global__ void k_init(float* __restrict__ xz, int nf, int* __restrict__ last) {
    int i = blockIdx.x * blockDim.x + threadIdx.x;
    int stride = gridDim.x * blockDim.x;
    for (int p = i; p < nf; p += stride) xz[p] = 0.f;
    if (i < 256) last[i] = -1;
}

// ---- masked embedding sum; scatter into reshaped f16 x; segment-max last[] ----
__global__ void __launch_bounds__(128) k_embed(
    const int* __restrict__ diag, const float* __restrict__ mask,
    const float* __restrict__ table, _Float16* __restrict__ x,
    int* __restrict__ last)
{
    __shared__ int   s_idx[C_];
    __shared__ float s_msk[C_];
    __shared__ int   s_has;
    int b  = blockIdx.x / V_;
    int vi = blockIdx.x % V_;
    int tid = threadIdx.x;
    if (tid == 0) s_has = 0;
    __syncthreads();
    if (tid < C_) {
        int base = (b * V_ + vi) * C_ + tid;
        s_idx[tid] = diag[base];
        float m = mask[base];
        s_msk[tid] = m;
        if (m > 0.f) atomicOr(&s_has, 1);
    }
    __syncthreads();
    float acc = 0.f;
#pragma unroll 8
    for (int c = 0; c < C_; ++c)
        acc += s_msk[c] * table[s_idx[c] * E_ + tid];
    int p = vi * E_ + tid;          // faithful .view reshape: flat idx in [0,12800)
    int m = p / H_;
    int v = p - m * H_;
    x[(b * E_ + m) * KP_ + v] = (_Float16)acc;
    if (tid == 0 && s_has) atomicMax(&last[b], vi);
}

// ---- pack W_ih / W_hh into WMMA f16 B-fragment layout (zero-padded) ----
__global__ void __launch_bounds__(32) k_frag(
    const float* __restrict__ Wih, const float* __restrict__ Whh,
    _Float16* __restrict__ fih, _Float16* __restrict__ fhh)
{
    int mat  = blockIdx.x / 112;
    int frag = blockIdx.x % 112;       // ((g*7+ht)*4 + kc)
    int l = threadIdx.x;
    int kc = frag & 3, jt = frag >> 2;
    int g = jt / 7, ht = jt % 7;
    int n = l & 15;
    int jcol = 16 * ht + n;
    int kb = 32 * kc + ((l >= 16) ? 8 : 0);
    const float* W = mat ? Whh : Wih;
    _Float16* dst = (mat ? fhh : fih) + (size_t)frag * 512 + l * 16;
#pragma unroll
    for (int hi = 0; hi < 16; ++hi) {
        int K = kb + ((hi < 8) ? hi : (hi + 8));   // K runs {kb..kb+7, kb+16..kb+23}
        float v = 0.f;
        if (jcol < H_ && K < H_) v = W[(g * H_ + jcol) * H_ + K];
        dst[hi] = (_Float16)v;
    }
}

// ---- parallel GEMM: Xg[t] = x_t @ W_ih^T + (b_ih + b_hh).
//      W_ih fragments staged once per block into LDS via the Tensor Data Mover;
//      B-fragment reads then come from LDS (ds_load_b128) instead of L2. ----
__global__ void __launch_bounds__(256) k_xgemm(
    const _Float16* __restrict__ x, const _Float16* __restrict__ fih,
    const float* __restrict__ b_ih, const float* __restrict__ b_hh,
    float* __restrict__ xg)
{
    extern __shared__ _Float16 sfih[];    // 112 frags * 512 halves = 114,688 B
    int t = blockIdx.x;
    int w = threadIdx.x >> 5, l = threadIdx.x & 31;

    if (threadIdx.x < 32)                 // wave 0 issues the DMA and waits TENSORcnt
        tdm_copy_to_lds(fih, sfih, (unsigned)(SZ_WF / 4));
    __syncthreads();

    int mt = w, n = l & 15;
    int kbs = (l >= 16) ? 8 : 0;
    AB a[4];
    const _Float16* xrow = x + (size_t)(t * E_ + 16 * mt + n) * KP_;
#pragma unroll
    for (int kc = 0; kc < 4; ++kc) {
        int kb = 32 * kc + kbs;
        a[kc].h[0] = *(const v8h*)(xrow + kb);
        a[kc].h[1] = *(const v8h*)(xrow + kb + 16);
    }
    for (int jt = 0; jt < NJT; ++jt) {
        int g = jt / 7, ht = jt % 7;
        int jcol = 16 * ht + n;
        float bias = 0.f;
        if (jcol < H_) bias = b_ih[g * H_ + jcol] + b_hh[g * H_ + jcol];
        v8f acc = v8f_splat(bias);
#pragma unroll
        for (int kc = 0; kc < 4; ++kc) {
            AB bf;
            const _Float16* bp = sfih + (size_t)(jt * 4 + kc) * 512 + l * 16;
            bf.h[0] = *(const v8h*)bp;
            bf.h[1] = *(const v8h*)(bp + 8);
            acc = __builtin_amdgcn_wmma_f32_16x16x32_f16(
                false, a[kc].v, false, bf.v, (short)0, acc, false, false);
        }
        float* dst = xg + ((size_t)(t * NMT + mt) * NJT + jt) * 256 + (size_t)l * 8;
        *(v8f*)dst = acc;
    }
}

// ---- sequential LSTM: 1 workgroup (28 waves), h double-buffered in LDS,
//      c-state and W_hh fragments persistent in VGPRs, 1 barrier/step ----
__global__ void __launch_bounds__(896) k_lstm(
    const float* __restrict__ xg, const _Float16* __restrict__ fhh,
    const int* __restrict__ last, const float* __restrict__ fcw,
    const float* __restrict__ fcb, float* __restrict__ out)
{
    extern __shared__ _Float16 hbuf[];   // [2][128][128] f16 = 64 KB
    const int tid = threadIdx.x, w = tid >> 5, l = tid & 31;
    const int ht = w % 7, pair = w / 7;  // wave owns col-tile ht, row-tiles {2*pair, 2*pair+1}
    const int n = l & 15;
    const int kbs = (l >= 16) ? 8 : 0;

    { // zero both h buffers (h0 = 0, and K-padding 112..127 stays 0 forever)
        float* z = (float*)hbuf;
        for (int i = tid; i < 16384; i += 896) z[i] = 0.f;
    }

    AB bf[4][4];                          // W_hh fragments: 4 gates x 4 K-chunks
#pragma unroll
    for (int g = 0; g < 4; ++g)
#pragma unroll
        for (int kc = 0; kc < 4; ++kc) {
            const _Float16* bp = fhh + ((size_t)((g * 7 + ht) * 4 + kc)) * 512 + l * 16;
            bf[g][kc].h[0] = *(const v8h*)bp;
            bf[g][kc].h[1] = *(const v8h*)(bp + 8);
        }

    v8f cst[2];
#pragma unroll
    for (int p = 0; p < 2; ++p) cst[p] = v8f_splat(0.f);

    __syncthreads();

    for (int t = 0; t < 256; ++t) {
        const _Float16* hb = hbuf + (size_t)(t & 1) * 16384;
        _Float16* hw = hbuf + (size_t)((t + 1) & 1) * 16384;

        // wave 0: emit output for step t-1 (h written last step, fenced by barrier)
        if (w == 0 && t > 0) {
            int col = last[t - 1]; if (col < 0) col += H_;
            float part = 0.f;
#pragma unroll
            for (int mm = l; mm < 128; mm += 32)
                part += fcw[mm] * (float)hb[mm * KP_ + col];
#pragma unroll
            for (int off = 16; off > 0; off >>= 1)
                part += __shfl_down(part, off, 32);
            if (l == 0) out[t - 1] = fsig(part + fcb[0]);
        }

#pragma unroll
        for (int p = 0; p < 2; ++p) {
            const int mt = 2 * pair + p;
            AB a[4];
            const _Float16* hr = hb + (size_t)(16 * mt + n) * KP_;
#pragma unroll
            for (int kc = 0; kc < 4; ++kc) {
                int kb = 32 * kc + kbs;
                a[kc].h[0] = *(const v8h*)(hr + kb);
                a[kc].h[1] = *(const v8h*)(hr + kb + 16);
            }
            const float* cbase = xg + (size_t)((t * NMT + mt) * NJT) * 256 + (size_t)l * 8;
            if (t + 1 < 256)
                __builtin_prefetch(xg + (size_t)(((t + 1) * NMT + mt) * NJT + ht) * 256 + (size_t)l * 8, 0, 1);

            v8f gacc[4];
#pragma unroll
            for (int g = 0; g < 4; ++g) {
                v8f acc = *(const v8f*)(cbase + (size_t)(g * 7 + ht) * 256);
#pragma unroll
                for (int kc = 0; kc < 4; ++kc)
                    acc = __builtin_amdgcn_wmma_f32_16x16x32_f16(
                        false, a[kc].v, false, bf[g][kc].v, (short)0, acc, false, false);
                gacc[g] = acc;
            }
            v8f cold = cst[p], cnew, hnew;
#pragma unroll
            for (int r = 0; r < 8; ++r) {
                float i_ = fsig(gacc[0][r]);
                float f_ = fsig(gacc[1][r]);
                float g_ = ftanh(gacc[2][r]);
                float o_ = fsig(gacc[3][r]);
                float c_ = f_ * cold[r] + i_ * g_;
                cnew[r] = c_;
                hnew[r] = o_ * ftanh(c_);
            }
            cst[p] = cnew;
            const int colg = 16 * ht + n;
            const int rbase = 16 * mt + 8 * (l >> 4);     // D layout: M = r + 8*(l/16)
#pragma unroll
            for (int r = 0; r < 8; ++r)
                hw[(rbase + r) * KP_ + colg] = (_Float16)hnew[r];
        }
        __syncthreads();
    }

    // final output (t = 255) — state landed in buffer 0 (256 & 1 == 0)
    if (w == 0) {
        const _Float16* hb = hbuf;
        int col = last[255]; if (col < 0) col += H_;
        float part = 0.f;
#pragma unroll
        for (int mm = l; mm < 128; mm += 32)
            part += fcw[mm] * (float)hb[mm * KP_ + col];
#pragma unroll
        for (int off = 16; off > 0; off >>= 1)
            part += __shfl_down(part, off, 32);
        if (l == 0) out[255] = fsig(part + fcb[0]);
    }
}

extern "C" void kernel_launch(void* const* d_in, const int* in_sizes, int n_in,
                              void* d_out, int out_size, void* d_ws, size_t ws_size,
                              hipStream_t stream) {
    const int*   diag  = (const int*)d_in[0];
    const float* mask  = (const float*)d_in[1];
    const float* table = (const float*)d_in[2];
    const float* W_ih  = (const float*)d_in[3];
    const float* W_hh  = (const float*)d_in[4];
    const float* b_ih  = (const float*)d_in[5];
    const float* b_hh  = (const float*)d_in[6];
    const float* fc_w  = (const float*)d_in[7];
    const float* fc_b  = (const float*)d_in[8];
    float* out = (float*)d_out;

    char* ws = (char*)d_ws;
    _Float16* x     = (_Float16*)(ws + OFF_X);
    _Float16* fih   = (_Float16*)(ws + OFF_WIH);
    _Float16* fhh   = (_Float16*)(ws + OFF_WHH);
    int*      lastb = (int*)(ws + OFF_LAST);
    float*    xg    = (float*)(ws + OFF_XG);

    k_init <<<512, 256, 0, stream>>>((float*)x, (int)(SZ_X / 4), lastb);
    k_embed<<<B_ * V_, 128, 0, stream>>>(diag, mask, table, x, lastb);
    k_frag <<<224, 32, 0, stream>>>(W_ih, W_hh, fih, fhh);
    k_xgemm<<<B_, 256, (unsigned)SZ_WF, stream>>>(x, fih, b_ih, b_hh, xg);
    k_lstm <<<1, 896, 65536, stream>>>(xg, fhh, lastb, fc_w, fc_b, out);
}
